// AnomalyAwareStorage_22617297781306
// MI455X (gfx1250) — compile-verified, matrix operand order checked by reading, not run on hardware
//
#include <hip/hip_runtime.h>

// ---------------------------------------------------------------------------
// AnomalyAwareStorage for MI455X (gfx1250, wave32, WMMA).
//
//  * Single fused kernel, 16 batch rows per workgroup (1024 WGs), 256 thr.
//  * One streaming pass over storage (non-temporal loads): masked S1/S2 stats
//    + decayed/rolled st written to global (non-temporal f32 stores) and
//    cached in LDS (bf16, 160 KB of the 320 KB WGP LDS).
//  * Attention algebraically refactored (single query):
//      scores[b,h,s] = (1/8) * <st[b,s,:], qw[b,h,:]>,  qw = q_h @ Wk_h
//      ctx[b,h,:]    = (sum_s attn * st[b,s,:]) @ Wv_h^T + bv
//    -> removes the two huge [B*K,256]x[256,256] GEMMs (20x fewer FLOPs).
//  * All dense GEMMs via V_WMMA_F32_16X16X32_BF16 (M=16 tiles).
//  * Weights pre-converted to bf16 (incl. Wk^T) in workspace; they stay hot
//    in L2 because all streaming traffic is marked non-temporal.
// ---------------------------------------------------------------------------

#define NB    16384
#define KSL   20
#define DD    256
#define NH    4

typedef __attribute__((ext_vector_type(16))) __bf16 v16bf;
typedef __attribute__((ext_vector_type(8)))  __bf16 v8bf;
typedef __attribute__((ext_vector_type(8)))  float  v8f;
typedef __attribute__((ext_vector_type(4)))  float  f32x4;

// ---- LDS layout (bytes) ----------------------------------------------------
#define SM_STBF    0                       // __bf16[16*20*256]  163840
#define SM_CTXBF   163840                  // __bf16[16*512]      16384
#define SM_NMF     180224                  // float [16*256]      16384
#define SM_QBF     196608                  // __bf16[16*256]       8192
#define SM_SCR     204800                  // 65536: h_ad@0 (8K), h_imp@8192 (16K); later qw@0 (64K)
#define SM_ATTNW   270336                  // float [16*4*20]      5120
#define SM_WSTBF   275456                  // __bf16[16*4*256]    32768
#define SM_CTXOBF  308224                  // __bf16[16*256]       8192
#define SM_SCAL    316416                  // float [32]            128
#define SMEM_TOTAL 316544                  // <= 320 KB WGP LDS

// ---- workspace layout (bf16 elements) --------------------------------------
#define WS_ADW1    0                       // 128*256
#define WS_IMPW1   32768                   // 256*512
#define WS_ATTNIN  163840                  // 768*256 (wq | wk | wv rows)
#define WS_WKT     360448                  // 256*256  wkT[j][dh] = wk[dh][j]
#define WS_ATTNOUT 425984                  // 256*256

// ---- WMMA helpers ----------------------------------------------------------
// A 16x32 bf16 fragment / B 32x16 fragment (B column n = weight row n):
// lane<16:  row=lane,    K = k0+{0..7}  and k0+{16..23}
// lane>=16: row=lane-16, K = k0+{8..15} and k0+{24..31}
__device__ __forceinline__ v16bf load_frag(const __bf16* base, int ld, int k0) {
  const int lane = threadIdx.x & 31;
  const int r    = lane & 15;
  const int sel  = (lane >> 4) & 1;
  const __bf16* p = base + (long)r * ld + k0 + sel * 8;
  union { v16bf v; v8bf h[2]; } u;
  u.h[0] = *(const v8bf*)(p);
  u.h[1] = *(const v8bf*)(p + 16);
  return u.v;
}

__device__ __forceinline__ v8f wmma_mac(const __bf16* A, int lda,
                                        const __bf16* Bp, int ldb,
                                        int K, v8f c) {
  for (int k0 = 0; k0 < K; k0 += 32) {
    v16bf a = load_frag(A,  lda, k0);
    v16bf b = load_frag(Bp, ldb, k0);
    c = __builtin_amdgcn_wmma_f32_16x16x32_bf16(false, a, false, b,
                                                (short)0, c, false, false);
  }
  return c;
}

// C/D tile: vgpr i, lane -> (m = i + 8*(lane>=16), n = lane&15)
__device__ __forceinline__ void store_c_f32(float* dst, int ldd, v8f c,
                                            const float* bias, bool do_relu) {
  const int lane = threadIdx.x & 31;
  const int n  = lane & 15;
  const int m0 = (lane >> 4) * 8;
  const float bb = bias ? bias[n] : 0.f;
#pragma unroll
  for (int i = 0; i < 8; ++i) {
    float v = c[i] + bb;
    if (do_relu) v = fmaxf(v, 0.f);
    dst[(long)(m0 + i) * ldd + n] = v;
  }
}

__device__ __forceinline__ void store_c_f32_nt(float* dst, int ldd, v8f c,
                                               const float* bias) {
  const int lane = threadIdx.x & 31;
  const int n  = lane & 15;
  const int m0 = (lane >> 4) * 8;
  const float bb = bias ? bias[n] : 0.f;
#pragma unroll
  for (int i = 0; i < 8; ++i)
    __builtin_nontemporal_store(c[i] + bb, dst + (long)(m0 + i) * ldd + n);
}

__device__ __forceinline__ void store_c_bf16(__bf16* dst, int ldd, v8f c,
                                             const float* bias) {
  const int lane = threadIdx.x & 31;
  const int n  = lane & 15;
  const int m0 = (lane >> 4) * 8;
  const float bb = bias ? bias[n] : 0.f;
#pragma unroll
  for (int i = 0; i < 8; ++i)
    dst[(long)(m0 + i) * ldd + n] = (__bf16)(c[i] + bb);
}

// ---- weight pre-conversion -------------------------------------------------
__global__ void cvt_bf16_kernel(const float* __restrict__ src,
                                __bf16* __restrict__ dst, int n) {
  int i = blockIdx.x * blockDim.x + threadIdx.x;
  if (i < n) dst[i] = (__bf16)src[i];
}

__global__ void cvt_bf16_T_kernel(const float* __restrict__ src,
                                  __bf16* __restrict__ dst, int rows, int cols) {
  int i = blockIdx.x * blockDim.x + threadIdx.x;
  if (i < rows * cols) {
    int r = i / cols, c = i % cols;
    dst[c * rows + r] = (__bf16)src[i];
  }
}

// ---- fused kernel ----------------------------------------------------------
__global__ __launch_bounds__(256, 1)
void anomaly_storage_fused(const float* __restrict__ storage,
                           const float* __restrict__ new_message,
                           const float* __restrict__ ad_b1,
                           const float* __restrict__ ad_w2,
                           const float* __restrict__ ad_b2,
                           const float* __restrict__ imp_b1,
                           const float* __restrict__ imp_w2,
                           const float* __restrict__ imp_b2,
                           const float* __restrict__ attn_in_b,
                           const float* __restrict__ attn_out_b,
                           const __bf16* __restrict__ adw1_bf,
                           const __bf16* __restrict__ impw1_bf,
                           const __bf16* __restrict__ attnin_bf,
                           const __bf16* __restrict__ wkT_bf,
                           const __bf16* __restrict__ attnout_bf,
                           float* __restrict__ st_out,
                           float* __restrict__ agg_out) {
  extern __shared__ unsigned char smem[];
  __bf16* st_bf   = (__bf16*)(smem + SM_STBF);
  __bf16* ctx_bf  = (__bf16*)(smem + SM_CTXBF);
  float*  nmf     = (float*) (smem + SM_NMF);
  __bf16* q_bf    = (__bf16*)(smem + SM_QBF);
  float*  h_ad    = (float*) (smem + SM_SCR);
  float*  h_imp   = (float*) (smem + SM_SCR + 8192);
  float*  qw      = (float*) (smem + SM_SCR);        // overlays h_ad/h_imp later
  float*  attnw   = (float*) (smem + SM_ATTNW);
  __bf16* wst_bf  = (__bf16*)(smem + SM_WSTBF);
  __bf16* ctxo_bf = (__bf16*)(smem + SM_CTXOBF);
  float*  statan  = (float*) (smem + SM_SCAL);
  float*  imp_s   = statan + 16;

  const int tid  = threadIdx.x;
  const int lane = tid & 31;
  const int wv   = tid >> 5;
  const int b0   = blockIdx.x * 16;
  const int d0   = lane * 8;

  // ---------------- Phase A: stats + decayed/rolled st (slots 1..19) -------
  for (int rr = 0; rr < 2; ++rr) {
    const int row = wv * 2 + rr;
    const long gb = b0 + row;
    float s1[8], s2[8];
#pragma unroll
    for (int i = 0; i < 8; ++i) { s1[i] = 0.f; s2[i] = 0.f; }
    float cnt = 0.f, dec = 1.f;
    const float* srow = storage + gb * (long)(KSL * DD) + d0;
    float*       orow = st_out  + gb * (long)(KSL * DD) + d0;

    for (int s = 0; s < KSL; ++s) {
      f32x4 x0 = __builtin_nontemporal_load((const f32x4*)(srow + (long)s * DD));
      f32x4 x1 = __builtin_nontemporal_load((const f32x4*)(srow + (long)s * DD + 4));
      float x[8] = {x0.x, x0.y, x0.z, x0.w, x1.x, x1.y, x1.z, x1.w};
      float asum = 0.f;
#pragma unroll
      for (int i = 0; i < 8; ++i) asum += fabsf(x[i]);
      for (int o = 16; o > 0; o >>= 1) asum += __shfl_xor(asum, o);
      const float m = (asum > 0.f) ? 1.f : 0.f;
      cnt += m;
#pragma unroll
      for (int i = 0; i < 8; ++i) { s1[i] += x[i] * m; s2[i] += x[i] * x[i] * m; }
      if (s < KSL - 1) {   // roll: st[:, s+1] = storage[:, s] * DECAY^s
        float y[8];
        v8bf  yv;
#pragma unroll
        for (int i = 0; i < 8; ++i) { y[i] = x[i] * dec; yv[i] = (__bf16)y[i]; }
        f32x4 y0 = {y[0], y[1], y[2], y[3]};
        f32x4 y1 = {y[4], y[5], y[6], y[7]};
        __builtin_nontemporal_store(y0, (f32x4*)(orow + (long)(s + 1) * DD));
        __builtin_nontemporal_store(y1, (f32x4*)(orow + (long)(s + 1) * DD + 4));
        *(v8bf*)(st_bf + (row * KSL + (s + 1)) * DD + d0) = yv;   // ds_store_b128
      }
      dec *= 0.95f;
    }

    const float denom = cnt + 1e-8f;
    f32x4 n0 = *(const f32x4*)(new_message + gb * (long)DD + d0);
    f32x4 n1 = *(const f32x4*)(new_message + gb * (long)DD + d0 + 4);
    float nm[8] = {n0.x, n0.y, n0.z, n0.w, n1.x, n1.y, n1.z, n1.w};
    int zc = 0;
    v8bf nmv, muv;
#pragma unroll
    for (int i = 0; i < 8; ++i) {
      const float mu   = s1[i] / denom;
      const float varn = fmaxf(s2[i] - 2.f * mu * s1[i] + mu * mu * cnt, 0.f);
      const float sd   = sqrtf(varn / denom);
      const float z    = fabsf((nm[i] - mu) / (sd + 1e-8f));
      zc += (z > 2.0f) ? 1 : 0;
      nmv[i] = (__bf16)nm[i];
      muv[i] = (__bf16)mu;
    }
    *(v8bf*)(ctx_bf + row * 512 + d0)        = nmv;
    *(v8bf*)(ctx_bf + row * 512 + 256 + d0)  = muv;
    *(f32x4*)(nmf + row * 256 + d0)          = n0;
    *(f32x4*)(nmf + row * 256 + d0 + 4)      = n1;
    for (int o = 16; o > 0; o >>= 1) zc += __shfl_xor(zc, o);
    if (lane == 0) statan[row] = (float)zc / 256.f;
  }
  __syncthreads();

  // ---------------- Phase B: WMMA ad-MLP, imp-MLP, q projection ------------
  {
    const int nt = wv;                                  // ad hidden: N=128
    v8f c = {};
    c = wmma_mac(ctx_bf, 512, adw1_bf + nt * 16 * 256, 256, 256, c);
    store_c_f32(h_ad + nt * 16, 128, c, ad_b1 + nt * 16, true);
  }
  for (int nt = wv; nt < 16; nt += 8) {                 // imp hidden: N=256,K=512
    v8f c = {};
    c = wmma_mac(ctx_bf, 512, impw1_bf + nt * 16 * 512, 512, 512, c);
    store_c_f32(h_imp + nt * 16, 256, c, imp_b1 + nt * 16, true);
  }
  for (int nt = wv; nt < 16; nt += 8) {                 // q = nm @ wq^T + bq
    v8f c = {};
    c = wmma_mac(ctx_bf, 512, attnin_bf + nt * 16 * 256, 256, 256, c);
    store_c_bf16(q_bf + nt * 16, 256, c, attn_in_b + nt * 16);
  }
  __syncthreads();

  // ---------------- Phase C1: per-row anomaly + importance scalar ----------
  if (tid < 16) {
    float sa = 0.f;
    for (int i = 0; i < 128; ++i) sa += h_ad[tid * 128 + i] * ad_w2[i];
    const float learned = 1.f / (1.f + expf(-(sa + ad_b2[0])));
    float si = 0.f;
    for (int i = 0; i < 256; ++i) si += h_imp[tid * 256 + i] * imp_w2[i];
    const float xx = si + imp_b2[0];
    const float sp = (xx > 20.f) ? xx : log1pf(expf(xx));
    imp_s[tid] = sp * (1.f + 0.5f * statan[tid] + 0.5f * learned);
  }
  __syncthreads();

  // ---------------- Phase C2: qw = q_h @ Wk_h (WMMA) + st slot 0 -----------
  for (int j = wv; j < 64; j += 8) {                    // 4 heads x 16 ntiles
    const int h = j >> 4, nt = j & 15;
    v8f c = {};
    c = wmma_mac(q_bf + h * 64, 256, wkT_bf + nt * 16 * 256 + h * 64, 256, 64, c);
    store_c_f32(qw + h * 256 + nt * 16, 1024, c, nullptr, false);
  }
  for (int idx = tid; idx < 16 * 256 / 8; idx += 256) { // st[:,0] = nm * imp
    const int b = idx >> 5, d8 = (idx & 31) * 8;
    const float im = imp_s[b];
    float y[8];
    v8bf  yv;
#pragma unroll
    for (int i = 0; i < 8; ++i) { y[i] = nmf[b * 256 + d8 + i] * im; yv[i] = (__bf16)y[i]; }
    f32x4 y0 = {y[0], y[1], y[2], y[3]};
    f32x4 y1 = {y[4], y[5], y[6], y[7]};
    float* o = st_out + ((long)(b0 + b) * KSL) * DD + d8;
    __builtin_nontemporal_store(y0, (f32x4*)(o));
    __builtin_nontemporal_store(y1, (f32x4*)(o + 4));
    *(v8bf*)(st_bf + (b * KSL) * DD + d8) = yv;
  }
  __syncthreads();

  // ---------------- Phase D: scores[b,h,s] = <st, qw> / 8 ------------------
  {
    const int p = tid >> 2;                    // (b,h) pair 0..63
    const int b = p >> 2, h = p & 3;
    const int s0 = (tid & 3) * 5;
    const float* qr = qw + (b * 4 + h) * 256;
    for (int s = s0; s < s0 + 5; ++s) {
      const v8bf* sr = (const v8bf*)(st_bf + (b * KSL + s) * DD);
      float acc = 0.f;
      for (int j8 = 0; j8 < 32; ++j8) {
        v8bf vv = sr[j8];
#pragma unroll
        for (int i = 0; i < 8; ++i) acc += (float)vv[i] * qr[j8 * 8 + i];
      }
      attnw[(b * 4 + h) * 20 + s] = acc * 0.125f;   // / sqrt(64)
    }
  }
  __syncthreads();

  // ---------------- Phase E: softmax over 20 slots per (b,h) ---------------
  if (tid < 64) {
    float* a = attnw + tid * 20;
    float mx = a[0];
    for (int s = 1; s < 20; ++s) mx = fmaxf(mx, a[s]);
    float sum = 0.f;
    for (int s = 0; s < 20; ++s) { float e = expf(a[s] - mx); a[s] = e; sum += e; }
    const float inv = 1.f / sum;
    for (int s = 0; s < 20; ++s) a[s] *= inv;
  }
  __syncthreads();

  // ---------------- Phase F: wst[b,h,:] = sum_s attn * st ------------------
  for (int idx = tid; idx < 16 * 4 * 256 / 8; idx += 256) {
    const int b = idx >> 7, h = (idx >> 5) & 3, j8 = (idx & 31) * 8;
    const float* a = attnw + (b * 4 + h) * 20;
    float acc[8];
#pragma unroll
    for (int i = 0; i < 8; ++i) acc[i] = 0.f;
    for (int s = 0; s < 20; ++s) {
      const float as = a[s];
      v8bf vv = *(const v8bf*)(st_bf + (b * KSL + s) * DD + j8);
#pragma unroll
      for (int i = 0; i < 8; ++i) acc[i] += as * (float)vv[i];
    }
    v8bf ov;
#pragma unroll
    for (int i = 0; i < 8; ++i) ov[i] = (__bf16)acc[i];
    *(v8bf*)(wst_bf + (b * 4 + h) * 256 + j8) = ov;
  }
  __syncthreads();

  // ---------------- Phase G: ctx_h = wst_h @ Wv_h^T + bv (WMMA) ------------
  for (int j = wv; j < 16; j += 8) {                    // 4 heads x 4 ntiles
    const int h = j >> 2, nt = j & 3;
    v8f c = {};
    c = wmma_mac(wst_bf + h * 256, 1024,
                 attnin_bf + (long)(512 + h * 64 + nt * 16) * 256, 256, 256, c);
    store_c_bf16(ctxo_bf + h * 64 + nt * 16, 256,
                 c, attn_in_b + 512 + h * 64 + nt * 16);
  }
  __syncthreads();

  // ---------------- Phase H: agg = ctx @ Wout^T + b (WMMA, NT stores) ------
  for (int nt = wv; nt < 16; nt += 8) {
    v8f c = {};
    c = wmma_mac(ctxo_bf, 256, attnout_bf + nt * 16 * 256, 256, 256, c);
    store_c_f32_nt(agg_out + (long)b0 * DD + nt * 16, DD, c, attn_out_b + nt * 16);
  }
}

// ---------------------------------------------------------------------------
extern "C" void kernel_launch(void* const* d_in, const int* in_sizes, int n_in,
                              void* d_out, int out_size, void* d_ws, size_t ws_size,
                              hipStream_t stream) {
  (void)in_sizes; (void)n_in; (void)out_size; (void)ws_size;
  const float* storage     = (const float*)d_in[0];
  const float* new_message = (const float*)d_in[1];
  const float* ad_w1       = (const float*)d_in[2];
  const float* ad_b1       = (const float*)d_in[3];
  const float* ad_w2       = (const float*)d_in[4];
  const float* ad_b2       = (const float*)d_in[5];
  const float* imp_w1      = (const float*)d_in[6];
  const float* imp_b1      = (const float*)d_in[7];
  const float* imp_w2      = (const float*)d_in[8];
  const float* imp_b2      = (const float*)d_in[9];
  const float* attn_in_w   = (const float*)d_in[10];
  const float* attn_in_b   = (const float*)d_in[11];
  const float* attn_out_w  = (const float*)d_in[12];
  const float* attn_out_b  = (const float*)d_in[13];

  __bf16* ws         = (__bf16*)d_ws;
  __bf16* adw1_bf    = ws + WS_ADW1;
  __bf16* impw1_bf   = ws + WS_IMPW1;
  __bf16* attnin_bf  = ws + WS_ATTNIN;
  __bf16* wkT_bf     = ws + WS_WKT;
  __bf16* attnout_bf = ws + WS_ATTNOUT;

  float* st_out  = (float*)d_out;
  float* agg_out = st_out + (long)NB * KSL * DD;

  // weight -> bf16 pre-conversion (stays L2-resident; streaming data is NT)
  cvt_bf16_kernel<<<(128 * 256 + 255) / 256, 256, 0, stream>>>(ad_w1, adw1_bf, 128 * 256);
  cvt_bf16_kernel<<<(256 * 512 + 255) / 256, 256, 0, stream>>>(imp_w1, impw1_bf, 256 * 512);
  cvt_bf16_kernel<<<(768 * 256 + 255) / 256, 256, 0, stream>>>(attn_in_w, attnin_bf, 768 * 256);
  cvt_bf16_T_kernel<<<(256 * 256 + 255) / 256, 256, 0, stream>>>(attn_in_w + 256 * 256, wkT_bf, 256, 256);
  cvt_bf16_kernel<<<(256 * 256 + 255) / 256, 256, 0, stream>>>(attn_out_w, attnout_bf, 256 * 256);

  (void)hipFuncSetAttribute(reinterpret_cast<const void*>(anomaly_storage_fused),
                            hipFuncAttributeMaxDynamicSharedMemorySize,
                            (int)SMEM_TOTAL);

  anomaly_storage_fused<<<NB / 16, 256, SMEM_TOTAL, stream>>>(
      storage, new_message, ad_b1, ad_w2, ad_b2, imp_b1, imp_w2, imp_b2,
      attn_in_b, attn_out_b, adw1_bf, impw1_bf, attnin_bf, wkT_bf, attnout_bf,
      st_out, agg_out);
}
